// GCN3D_Feb13_twoLayer_25520695673031
// MI455X (gfx1250) — compile-verified, matrix operand order checked by reading, not run on hardware
//
#include <hip/hip_runtime.h>
#include <math.h>

typedef __attribute__((ext_vector_type(16))) __bf16 v16bf;
typedef __attribute__((ext_vector_type(8)))  float  v8f;

#define F1 64
#define F2 3

// ---------------- utility kernels ----------------

__global__ void zero_i32(int* p, long long n) {
  long long i = (long long)blockIdx.x * blockDim.x + threadIdx.x;
  long long stride = (long long)gridDim.x * blockDim.x;
  for (; i < n; i += stride) p[i] = 0;
}

// int degree histogram over dst
__global__ void deg_count(const long long* __restrict__ dst, int* __restrict__ degi, int E) {
  int e = blockIdx.x * blockDim.x + threadIdx.x;
  if (e < E) atomicAdd(&degi[dst[e]], 1);
}

__global__ void make_dinv(const int* __restrict__ degi, float* __restrict__ dinv, int n) {
  int i = blockIdx.x * blockDim.x + threadIdx.x;
  if (i < n) dinv[i] = rsqrtf((float)degi[i] + 1.0f);
}

// ---- exclusive scan of degi -> rowstart (3 kernels: block scan, scan of sums, add) ----

__global__ void scan_block(const int* __restrict__ degi, int* __restrict__ rowstart,
                           int* __restrict__ blockSums, int n) {
  __shared__ int tmp[256];
  int i = blockIdx.x * 256 + threadIdx.x;
  int v = (i < n) ? degi[i] : 0;
  tmp[threadIdx.x] = v;
  __syncthreads();
#pragma unroll
  for (int off = 1; off < 256; off <<= 1) {
    int t = (threadIdx.x >= off) ? tmp[threadIdx.x - off] : 0;
    __syncthreads();
    tmp[threadIdx.x] += t;
    __syncthreads();
  }
  if (i < n) rowstart[i] = tmp[threadIdx.x] - v;   // exclusive
  if (threadIdx.x == 255) blockSums[blockIdx.x] = tmp[255];
}

__global__ void scan_sums(int* blockSums, int nb) {
  if (blockIdx.x == 0 && threadIdx.x == 0) {
    int run = 0;
    for (int b = 0; b < nb; ++b) { int t = blockSums[b]; blockSums[b] = run; run += t; }
  }
}

__global__ void scan_fixup(int* __restrict__ rowstart, const int* __restrict__ blockSums,
                           int* __restrict__ cursor, int n) {
  int i = blockIdx.x * 256 + threadIdx.x;
  if (i < n) {
    int rs = rowstart[i] + blockSums[i >> 8];
    rowstart[i] = rs;
    cursor[i] = rs;
  }
}

// bucket edges by dst: esrc[pos] = src node, ewt[pos] = dinv[s]*dinv[d]
__global__ void fill_csr(const long long* __restrict__ src, const long long* __restrict__ dst,
                         const float* __restrict__ dinv, int* __restrict__ cursor,
                         int* __restrict__ esrc, float* __restrict__ ewt, int E) {
  int e = blockIdx.x * blockDim.x + threadIdx.x;
  if (e >= E) return;
  int s = (int)src[e], d = (int)dst[e];
  int pos = atomicAdd(&cursor[d], 1);
  esrc[pos] = s;
  ewt[pos]  = dinv[s] * dinv[d];
}

// ---------------- pack W1 [K,64] f32 into bf16 B-operand panels ----------------
// For tile (kb,nt), lane L loads 16 contiguous bf16 (32B):
//   col n = nt*16 + (L&15), kbase = (L&16)?8:0
//   j<8  -> k = kb*32 + kbase + j ; j>=8 -> k = kb*32 + kbase + 16 + (j-8)
__global__ void pack_w1_bf16(const float* __restrict__ W1, __bf16* __restrict__ Bp, int K) {
  int t = blockIdx.x * blockDim.x + threadIdx.x;
  int j  = t & 15;
  int L  = (t >> 4) & 31;
  int nt = (t >> 9) & 3;
  int kb = t >> 11;
  if (kb * 32 >= K) return;
  int n = nt * 16 + (L & 15);
  int kbase = (L & 16) ? 8 : 0;
  int k = kb * 32 + kbase + (j < 8 ? j : 16 + (j - 8));
  Bp[t] = (__bf16)W1[(size_t)k * F1 + n];
}

// ---------------- layer-1 GEMM: h1 = x @ W1 ----------------
// 256 threads = 8 waves; each wave owns a 16x64 output row-panel (4 accumulators)
// so every byte of x is loaded exactly once. B panel (K*64 bf16 = 128KB) is staged
// into LDS with global_load_async_to_lds_b128 and read back with ds loads.
__global__ void __launch_bounds__(256)
gemm1_wmma(const float* __restrict__ x, const __bf16* __restrict__ Bp,
           float* __restrict__ h1, int Nrows, int K) {
  extern __shared__ __bf16 Bs[];          // K * 64 bf16
  const int tid  = threadIdx.x;
  const int lane = tid & 31;
  const int wv   = tid >> 5;              // 0..7

  // ---- async-stage B panel into LDS (16B per thread per iteration) ----
  {
    const int totalBytes = K * F1 * 2;    // 128KB for K=1024
    unsigned ldsBase = (unsigned)(size_t)&Bs[0];
    unsigned long long gBase = (unsigned long long)(size_t)Bp;
    for (int it = tid; it * 16 < totalBytes; it += 256) {
      unsigned      loff = ldsBase + (unsigned)(it * 16);
      unsigned long long ga = gBase + (unsigned long long)(it * 16);
      asm volatile("global_load_async_to_lds_b128 %0, %1, off"
                   :: "v"(loff), "v"(ga) : "memory");
    }
    asm volatile("s_wait_asynccnt 0" ::: "memory");
    __syncthreads();
  }

  int rowBase = blockIdx.x * 128 + wv * 16;
  int mrow = rowBase + (lane & 15);
  if (mrow >= Nrows) mrow = Nrows - 1;    // clamp loads; stores guarded below
  int kbase = (lane & 16) ? 8 : 0;
  const float* xr = x + (size_t)mrow * K;
  const v16bf* BsV = (const v16bf*)Bs;

  v8f c0 = {}, c1 = {}, c2 = {}, c3 = {};
  int nkb = K >> 5;
  for (int kb = 0; kb < nkb; ++kb) {
    int k0 = kb * 32 + kbase;
    float4 q0 = *(const float4*)(xr + k0);
    float4 q1 = *(const float4*)(xr + k0 + 4);
    float4 r0 = *(const float4*)(xr + k0 + 16);
    float4 r1 = *(const float4*)(xr + k0 + 20);
    v16bf a;
    a.s0=(__bf16)q0.x; a.s1=(__bf16)q0.y; a.s2=(__bf16)q0.z; a.s3=(__bf16)q0.w;
    a.s4=(__bf16)q1.x; a.s5=(__bf16)q1.y; a.s6=(__bf16)q1.z; a.s7=(__bf16)q1.w;
    a.s8=(__bf16)r0.x; a.s9=(__bf16)r0.y; a.sa=(__bf16)r0.z; a.sb=(__bf16)r0.w;
    a.sc=(__bf16)r1.x; a.sd=(__bf16)r1.y; a.se=(__bf16)r1.z; a.sf=(__bf16)r1.w;

    size_t bbase = ((size_t)kb * 4) * 32 + lane;
    v16bf b0 = BsV[bbase + 0 * 32];
    v16bf b1v = BsV[bbase + 1 * 32];
    v16bf b2v = BsV[bbase + 2 * 32];
    v16bf b3v = BsV[bbase + 3 * 32];

    c0 = __builtin_amdgcn_wmma_f32_16x16x32_bf16(false, a, false, b0,  (short)0, c0, false, false);
    c1 = __builtin_amdgcn_wmma_f32_16x16x32_bf16(false, a, false, b1v, (short)0, c1, false, false);
    c2 = __builtin_amdgcn_wmma_f32_16x16x32_bf16(false, a, false, b2v, (short)0, c2, false, false);
    c3 = __builtin_amdgcn_wmma_f32_16x16x32_bf16(false, a, false, b3v, (short)0, c3, false, false);
  }

  // C/D layout: VGPR r, lanes 0-15 -> M=r, lanes 16-31 -> M=r+8; N = lane&15
  int colBase = lane & 15;
  int rowOff  = (lane & 16) ? 8 : 0;
  v8f acc[4] = {c0, c1, c2, c3};
#pragma unroll
  for (int nt = 0; nt < 4; ++nt) {
#pragma unroll
    for (int r = 0; r < 8; ++r) {
      int row = rowBase + rowOff + r;
      if (row < Nrows) h1[(size_t)row * F1 + nt * 16 + colBase] = acc[nt][r];
    }
  }
}

// ---------------- layer-1 aggregate (CSR gather) + self-loop + ELU + GEMM2 ----------------
// One wave32 per node: 2 features per lane, in-edge loop, shuffle reduce for 3 cols.
__global__ void __launch_bounds__(256)
layer1_agg(const int* __restrict__ rowstart, const int* __restrict__ degi,
           const int* __restrict__ esrc, const float* __restrict__ ewt,
           const float* __restrict__ h1, const float* __restrict__ dinv,
           const float* __restrict__ b1, const float* __restrict__ W2,
           float* __restrict__ h2, int Nrows) {
  int lane = threadIdx.x & 31;
  int node = blockIdx.x * 8 + (threadIdx.x >> 5);
  if (node >= Nrows) return;
  int start = rowstart[node];
  int cnt   = degi[node];
  float acc0 = 0.f, acc1 = 0.f;
  for (int j = 0; j < cnt; ++j) {
    int   s = esrc[start + j];     // uniform across wave -> broadcast
    float w = ewt[start + j];
    const float* hr = h1 + (size_t)s * F1;
    acc0 += hr[lane]      * w;     // h1 is L2-resident (25.6MB < 192MB)
    acc1 += hr[lane + 32] * w;
  }
  float di = dinv[node];
  float self = di * di;
  const float* hn = h1 + (size_t)node * F1;
  float v0 = acc0 + hn[lane]      * self + b1[lane];
  float v1 = acc1 + hn[lane + 32] * self + b1[lane + 32];
  v0 = (v0 > 0.0f) ? v0 : (expf(v0) - 1.0f);
  v1 = (v1 > 0.0f) ? v1 : (expf(v1) - 1.0f);
  float p0 = v0 * W2[lane * F2 + 0] + v1 * W2[(lane + 32) * F2 + 0];
  float p1 = v0 * W2[lane * F2 + 1] + v1 * W2[(lane + 32) * F2 + 1];
  float p2 = v0 * W2[lane * F2 + 2] + v1 * W2[(lane + 32) * F2 + 2];
#pragma unroll
  for (int off = 16; off > 0; off >>= 1) {
    p0 += __shfl_xor(p0, off, 32);
    p1 += __shfl_xor(p1, off, 32);
    p2 += __shfl_xor(p2, off, 32);
  }
  if (lane == 0) {
    h2[(size_t)node * F2 + 0] = p0;
    h2[(size_t)node * F2 + 1] = p1;
    h2[(size_t)node * F2 + 2] = p2;
  }
}

// ---------------- layer-2 aggregate (CSR gather) + self-loop + b2 + ELU -> out ----------------
// One wave32 per node: lanes parallel over in-edges, shuffle reduce 3 sums.
__global__ void __launch_bounds__(256)
layer2_agg(const int* __restrict__ rowstart, const int* __restrict__ degi,
           const int* __restrict__ esrc, const float* __restrict__ ewt,
           const float* __restrict__ h2, const float* __restrict__ dinv,
           const float* __restrict__ b2, float* __restrict__ out, int Nrows) {
  int lane = threadIdx.x & 31;
  int node = blockIdx.x * 8 + (threadIdx.x >> 5);
  if (node >= Nrows) return;
  int start = rowstart[node];
  int cnt   = degi[node];
  float s0 = 0.f, s1 = 0.f, s2 = 0.f;
  for (int j = lane; j < cnt; j += 32) {
    int   s = esrc[start + j];
    float w = ewt[start + j];
    const float* hr = h2 + (size_t)s * F2;
    s0 += hr[0] * w; s1 += hr[1] * w; s2 += hr[2] * w;
  }
#pragma unroll
  for (int off = 16; off > 0; off >>= 1) {
    s0 += __shfl_xor(s0, off, 32);
    s1 += __shfl_xor(s1, off, 32);
    s2 += __shfl_xor(s2, off, 32);
  }
  if (lane == 0) {
    float di = dinv[node];
    float self = di * di;
    const float* hn = h2 + (size_t)node * F2;
    float o0 = s0 + hn[0] * self + b2[0];
    float o1 = s1 + hn[1] * self + b2[1];
    float o2 = s2 + hn[2] * self + b2[2];
    o0 = (o0 > 0.0f) ? o0 : (expf(o0) - 1.0f);
    o1 = (o1 > 0.0f) ? o1 : (expf(o1) - 1.0f);
    o2 = (o2 > 0.0f) ? o2 : (expf(o2) - 1.0f);
    out[(size_t)node * F2 + 0] = o0;
    out[(size_t)node * F2 + 1] = o1;
    out[(size_t)node * F2 + 2] = o2;
  }
}

// ---------------- launcher ----------------
extern "C" void kernel_launch(void* const* d_in, const int* in_sizes, int n_in,
                              void* d_out, int out_size, void* d_ws, size_t ws_size,
                              hipStream_t stream) {
  const float*     x   = (const float*)d_in[0];
  const long long* adj = (const long long*)d_in[1];   // int64 [2, E]
  const float*     W1  = (const float*)d_in[5];
  const float*     b1  = (const float*)d_in[6];
  const float*     W2  = (const float*)d_in[7];
  const float*     b2  = (const float*)d_in[8];

  const int Nrows = in_sizes[3];            // in_batch has N elements
  const int E     = in_sizes[1] / 2;
  const int K     = in_sizes[0] / Nrows;    // 1024
  const int nb    = (Nrows + 255) / 256;    // scan blocks

  // workspace layout (256B aligned slabs)
  char* ws = (char*)d_ws;
  size_t off = 0;
  auto alloc = [&](size_t bytes) -> void* {
    void* p = ws + off;
    off += (bytes + 255) & ~(size_t)255;
    return p;
  };
  int*    degi     = (int*)  alloc((size_t)Nrows * sizeof(int));
  float*  dinv     = (float*)alloc((size_t)Nrows * sizeof(float));
  int*    rowstart = (int*)  alloc((size_t)Nrows * sizeof(int));
  int*    cursor   = (int*)  alloc((size_t)Nrows * sizeof(int));
  int*    blockSums= (int*)  alloc((size_t)nb    * sizeof(int));
  int*    esrc     = (int*)  alloc((size_t)E     * sizeof(int));
  float*  ewt      = (float*)alloc((size_t)E     * sizeof(float));
  float*  h1       = (float*)alloc((size_t)Nrows * F1 * sizeof(float));
  float*  h2       = (float*)alloc((size_t)Nrows * F2 * sizeof(float));
  __bf16* Bp       = (__bf16*)alloc((size_t)K * F1 * sizeof(__bf16));
  (void)ws_size; (void)n_in; (void)out_size;

  const long long* src  = adj;
  const long long* dstp = adj + E;

  // degree histogram -> dinv
  zero_i32<<<512, 256, 0, stream>>>(degi, (long long)Nrows);
  deg_count<<<(E + 255) / 256, 256, 0, stream>>>(dstp, degi, E);
  make_dinv<<<(Nrows + 255) / 256, 256, 0, stream>>>(degi, dinv, Nrows);

  // CSR build: exclusive scan + edge bucketing
  scan_block<<<nb, 256, 0, stream>>>(degi, rowstart, blockSums, Nrows);
  scan_sums<<<1, 1, 0, stream>>>(blockSums, nb);
  scan_fixup<<<nb, 256, 0, stream>>>(rowstart, blockSums, cursor, Nrows);
  fill_csr<<<(E + 255) / 256, 256, 0, stream>>>(src, dstp, dinv, cursor, esrc, ewt, E);

  // W1 -> bf16 B panels
  int packElems = K * F1;
  pack_w1_bf16<<<(packElems + 255) / 256, 256, 0, stream>>>(W1, Bp, K);

  // h1 = x @ W1  (WMMA; B staged to LDS via async loads)
  size_t ldsBytes = (size_t)K * F1 * sizeof(__bf16);   // 128KB
  gemm1_wmma<<<(Nrows + 127) / 128, 256, ldsBytes, stream>>>(x, Bp, h1, Nrows, K);

  // layer 1 aggregate + ELU + GEMM2 -> h2
  layer1_agg<<<(Nrows + 7) / 8, 256, 0, stream>>>(rowstart, degi, esrc, ewt,
                                                  h1, dinv, b1, W2, h2, Nrows);

  // layer 2 aggregate + ELU -> out
  layer2_agg<<<(Nrows + 7) / 8, 256, 0, stream>>>(rowstart, degi, esrc, ewt,
                                                  h2, dinv, b2, (float*)d_out, Nrows);
}
